// Head2_42477226557492
// MI455X (gfx1250) — matrix operand order repaired and checked by reading, not compiled, not run on hardware
//
#include <hip/hip_runtime.h>
#include <hip/hip_bf16.h>

#define BATCH 32
#define EMB 128
#define HID 32            // EMB / 4
#define HWT 8400
#define NIDX 100
#define NT 80             // spatial columns per block (divides 6400/1600/400)
#define NSUB 5            // NT / 16

typedef __attribute__((ext_vector_type(8)))  float  v8f;
typedef __attribute__((ext_vector_type(16))) __bf16 v16bf;

union BF16x16 {
    v16bf v;
    uint4 q[2];
    unsigned short u[16];
};

static __device__ __forceinline__ unsigned short f32_to_bf16_rne(float f) {
    unsigned int u = __float_as_uint(f);
    unsigned int r = 0x7FFFu + ((u >> 16) & 1u);
    u += r;
    return (unsigned short)(u >> 16);
}

static __device__ __forceinline__ unsigned int pack_bf16x2(float lo, float hi) {
    return (unsigned int)f32_to_bf16_rne(lo) |
           ((unsigned int)f32_to_bf16_rne(hi) << 16);
}

static __device__ __forceinline__ float sigmoidf_dev(float x) {
    return 1.0f / (1.0f + __expf(-x));
}

// ---------------------------------------------------------------------------
// Weight f32 -> bf16 conversion
// ---------------------------------------------------------------------------
__global__ void cvt_bf16_kernel(const float* __restrict__ src,
                                unsigned short* __restrict__ dst, int n) {
    int i = blockIdx.x * blockDim.x + threadIdx.x;
    if (i < n) dst[i] = f32_to_bf16_rne(src[i]);
}

// ---------------------------------------------------------------------------
// Neck 1x1 conv as WMMA GEMM:
//   feat[b, o, hwBase+hw] = sum_c W[o,c]*x[b,c,hw] + bias[o]
// Block: 256 threads = 8 waves; wave w owns M-rows [16w,16w+16) of a
// 128 x 80 output strip (5 N-subtiles, 5 WMMAs per 32-K chunk, shared A frag).
// Double-buffered LDS (one barrier per K-chunk): next chunk's global loads
// overlap the current chunk's WMMA burst. Staging packs bf16 pairs -> b32.
// ---------------------------------------------------------------------------
__global__ void __launch_bounds__(256)
neck_wmma_kernel(const float* __restrict__ x,              // [B, Cin, HWl]
                 const unsigned short* __restrict__ wbf,   // [128, Cin] bf16
                 const float* __restrict__ bias,           // [128]
                 float* __restrict__ feat,                 // [B, 128, 8400]
                 int Cin, int HWl, int hwBase) {
    const int tid  = threadIdx.x;
    const int lane = tid & 31;
    const int wid  = tid >> 5;               // 0..7
    const int hw0  = blockIdx.x * NT;
    const int b    = blockIdx.y;

    // [buf][n][k] bf16; row stride 48 ushorts (96B) keeps rows 32B-aligned
    __shared__ __align__(32) unsigned short ldsB[2][NT][48];

    const float* xb = x + (size_t)b * (size_t)Cin * (size_t)HWl;

    // WMMA fragment lane mapping
    const int m0   = wid * 16;
    const int arow = m0 + (lane & 15);
    const int kbA  = (lane < 16) ? 0 : 8;    // A: K base within chunk
    const int kbB  = (lane < 16) ? 0 : 8;    // B: K base within chunk
    const int ncol = lane & 15;

    const unsigned short* ap_base = wbf + (size_t)arow * Cin + kbA;

    // staging lane mapping: pair p = tid + 256*i -> column n, k-pair 2*cp
    // (32 k x 80 n tile = 1280 bf16 pairs, 5 pairs per thread)
    const int np[5] = { (tid + 0)   % NT, (tid + 256) % NT, (tid + 512) % NT,
                        (tid + 768) % NT, (tid + 1024) % NT };
    const int cp[5] = { (tid + 0)   / NT, (tid + 256) / NT, (tid + 512) / NT,
                        (tid + 768) / NT, (tid + 1024) / NT };

    v8f acc[NSUB];
#pragma unroll
    for (int s = 0; s < NSUB; ++s) acc[s] = (v8f){};

    // ---- prologue: stage chunk 0 into buffer 0 ----
    {
        float lo[5], hi[5];
#pragma unroll
        for (int i = 0; i < 5; ++i) {
            lo[i] = xb[(size_t)(2 * cp[i])     * HWl + hw0 + np[i]];
            hi[i] = xb[(size_t)(2 * cp[i] + 1) * HWl + hw0 + np[i]];
        }
#pragma unroll
        for (int i = 0; i < 5; ++i)
            *reinterpret_cast<unsigned int*>(&ldsB[0][np[i]][2 * cp[i]]) =
                pack_bf16x2(lo[i], hi[i]);
    }
    __syncthreads();

    int cur = 0;
    for (int kc = 0; kc < Cin; kc += 32) {
        const bool more = (kc + 32) < Cin;

        // --- issue next chunk's global loads (overlap with WMMA burst) ---
        float lo[5], hi[5];
        if (more) {
#pragma unroll
            for (int i = 0; i < 5; ++i) {
                lo[i] = xb[(size_t)(kc + 32 + 2 * cp[i])     * HWl + hw0 + np[i]];
                hi[i] = xb[(size_t)(kc + 32 + 2 * cp[i] + 1) * HWl + hw0 + np[i]];
            }
        }

        // --- A fragment: row 'arow', K = kc+kbA+{0..7} and +16+{0..7} ---
        BF16x16 a;
        const unsigned short* ap = ap_base + kc;
        a.q[0] = *reinterpret_cast<const uint4*>(ap);
        a.q[1] = *reinterpret_cast<const uint4*>(ap + 16);

        // --- 5 B fragments from LDS + 5 WMMAs sharing the A fragment ---
        BF16x16 bm[NSUB];
#pragma unroll
        for (int s = 0; s < NSUB; ++s) {
            const unsigned short* bp = &ldsB[cur][s * 16 + ncol][kbB];
            bm[s].q[0] = *reinterpret_cast<const uint4*>(bp);
            bm[s].q[1] = *reinterpret_cast<const uint4*>(bp + 16);
        }
#pragma unroll
        for (int s = 0; s < NSUB; ++s) {
            acc[s] = __builtin_amdgcn_wmma_f32_16x16x32_bf16(
                false, a.v, false, bm[s].v, (short)0, acc[s], false, false);
        }

        // --- stage next chunk into the other buffer; single barrier ---
        if (more) {
#pragma unroll
            for (int i = 0; i < 5; ++i)
                *reinterpret_cast<unsigned int*>(&ldsB[cur ^ 1][np[i]][2 * cp[i]]) =
                    pack_bf16x2(lo[i], hi[i]);
            __syncthreads();
            cur ^= 1;
        }
    }

    // D layout: VGPR r -> M = m0 + r (lanes 0-15) / m0 + 8 + r (lanes 16-31)
    const int mbase = m0 + ((lane >> 4) << 3);
#pragma unroll
    for (int s = 0; s < NSUB; ++s) {
        float* fout = feat + (size_t)b * EMB * HWT
                    + (size_t)(hwBase + hw0 + s * 16 + ncol);
#pragma unroll
        for (int r = 0; r < 8; ++r) {
            int o = mbase + r;
            fout[(size_t)o * HWT] = acc[s][r] + bias[o];
        }
    }
}

// ---------------------------------------------------------------------------
// Per-(level,b,c) channel max/mean over the level's spatial extent
// ---------------------------------------------------------------------------
__global__ void chan_stats_kernel(const float* __restrict__ feat,
                                  float* __restrict__ mx,   // [3][B][128]
                                  float* __restrict__ av) { // [3][B][128]
    const int c = blockIdx.x, b = blockIdx.y, l = blockIdx.z;
    const int hwBase = (l == 0) ? 0 : (l == 1) ? 6400 : 8000;
    const int hwLen  = (l == 0) ? 6400 : (l == 1) ? 1600 : 400;
    const float* p = feat + ((size_t)b * EMB + c) * HWT + hwBase;

    float m = -3.402823466e38f, s = 0.0f;
    for (int i = threadIdx.x; i < hwLen; i += blockDim.x) {
        float v = p[i];
        m = fmaxf(m, v);
        s += v;
    }
    __shared__ float sm[256], ss[256];
    sm[threadIdx.x] = m;
    ss[threadIdx.x] = s;
    __syncthreads();
    for (int st = 128; st > 0; st >>= 1) {
        if ((int)threadIdx.x < st) {
            sm[threadIdx.x] = fmaxf(sm[threadIdx.x], sm[threadIdx.x + st]);
            ss[threadIdx.x] += ss[threadIdx.x + st];
        }
        __syncthreads();
    }
    if (threadIdx.x == 0) {
        size_t o = ((size_t)l * BATCH + b) * EMB + c;
        mx[o] = sm[0];
        av[o] = ss[0] / (float)hwLen;
    }
}

// ---------------------------------------------------------------------------
// CAM gate: sigmoid(mlp(mx) + mlp(av)) for one level; one block per batch
// ---------------------------------------------------------------------------
__global__ void cam_gate_kernel(const float* __restrict__ mx,   // [B][128]
                                const float* __restrict__ av,   // [B][128]
                                const float* __restrict__ w1,   // [32][128]
                                const float* __restrict__ b1,   // [32]
                                const float* __restrict__ w2,   // [128][32]
                                const float* __restrict__ b2,   // [128]
                                float* __restrict__ gate) {     // [B][128]
    const int b = blockIdx.x, t = threadIdx.x;
    __shared__ float smx[EMB], sav[EMB], hm[HID], ha[HID];
    smx[t] = mx[(size_t)b * EMB + t];
    sav[t] = av[(size_t)b * EMB + t];
    __syncthreads();
    if (t < HID) {
        float s1 = b1[t], s2 = b1[t];
        for (int c = 0; c < EMB; ++c) {
            float w = w1[t * EMB + c];
            s1 += w * smx[c];
            s2 += w * sav[c];
        }
        hm[t] = fmaxf(s1, 0.0f);
        ha[t] = fmaxf(s2, 0.0f);
    }
    __syncthreads();
    float o = 2.0f * b2[t];
    for (int j = 0; j < HID; ++j) o += w2[t * HID + j] * (hm[j] + ha[j]);
    gate[(size_t)b * EMB + t] = sigmoidf_dev(o);
}

// ---------------------------------------------------------------------------
// Spatial max/mean over channels of gate*feat (gated map never materialized)
// ---------------------------------------------------------------------------
__global__ void spatial_stats_kernel(const float* __restrict__ feat,
                                     const float* __restrict__ gate, // [3][B][128]
                                     float* __restrict__ smx,        // [B][8400]
                                     float* __restrict__ sav) {      // [B][8400]
    int idx = blockIdx.x * blockDim.x + threadIdx.x;
    if (idx >= BATCH * HWT) return;
    int b = idx / HWT, hw = idx % HWT;
    int l = (hw < 6400) ? 0 : (hw < 8000) ? 1 : 2;
    const float* g = gate + ((size_t)l * BATCH + b) * EMB;
    const float* p = feat + (size_t)b * EMB * HWT + hw;
    float m = -3.402823466e38f, s = 0.0f;
    for (int c = 0; c < EMB; ++c) {
        float v = g[c] * p[(size_t)c * HWT];
        m = fmaxf(m, v);
        s += v;
    }
    smx[idx] = m;
    sav[idx] = s * (1.0f / EMB);
}

// ---------------------------------------------------------------------------
// SAM: 7x7 conv over [smx; sav] with pad 3, then sigmoid -> att[b][hw]
// ---------------------------------------------------------------------------
__global__ void sam_conv_kernel(const float* __restrict__ smx,
                                const float* __restrict__ sav,
                                const float* __restrict__ sw,  // [1][2][7][7]
                                float* __restrict__ att, int H, int W, int hwBase) {
    int hwl = H * W;
    int idx = blockIdx.x * blockDim.x + threadIdx.x;
    if (idx >= BATCH * hwl) return;
    int b = idx / hwl, p = idx % hwl;
    int y = p / W, x = p % W;
    const float* mb = smx + (size_t)b * HWT + hwBase;
    const float* ab = sav + (size_t)b * HWT + hwBase;
    float s = 0.0f;
    for (int ky = 0; ky < 7; ++ky) {
        int yy = y + ky - 3;
        if (yy < 0 || yy >= H) continue;
        for (int kx = 0; kx < 7; ++kx) {
            int xx = x + kx - 3;
            if (xx < 0 || xx >= W) continue;
            int q = yy * W + xx;
            s += sw[ky * 7 + kx] * mb[q] + sw[49 + ky * 7 + kx] * ab[q];
        }
    }
    att[(size_t)b * HWT + hwBase + p] = sigmoidf_dev(s);
}

// ---------------------------------------------------------------------------
// Gather: emb[b,k,c] = feat[b,c,j] * (1 + att[b,j]*gate[l(j)][b,c])
// ---------------------------------------------------------------------------
__global__ void gather_emb_kernel(const float* __restrict__ feat,
                                  const float* __restrict__ gate,
                                  const float* __restrict__ att,
                                  const int* __restrict__ reid_idx,
                                  float* __restrict__ out) {
    int bk = blockIdx.x;           // 0 .. B*NIDX-1
    int c  = threadIdx.x;          // 0 .. 127
    int b  = bk / NIDX;
    int j  = reid_idx[bk];
    int l  = (j < 6400) ? 0 : (j < 8000) ? 1 : 2;
    float g = gate[((size_t)l * BATCH + b) * EMB + c];
    float a = att[(size_t)b * HWT + j];
    float v = feat[((size_t)b * EMB + c) * HWT + j];
    out[(size_t)bk * EMB + c] = v * (1.0f + a * g);
}

// ---------------------------------------------------------------------------
extern "C" void kernel_launch(void* const* d_in, const int* in_sizes, int n_in,
                              void* d_out, int out_size, void* d_ws, size_t ws_size,
                              hipStream_t stream) {
    (void)in_sizes; (void)n_in; (void)out_size; (void)ws_size;

    const float* xin0 = (const float*)d_in[0];
    const float* xin1 = (const float*)d_in[1];
    const float* xin2 = (const float*)d_in[2];
    const float* yolo = (const float*)d_in[3];
    const int*   ridx = (const int*)d_in[4];
    const float* nw0  = (const float*)d_in[5];
    const float* nb0  = (const float*)d_in[6];
    const float* nw1  = (const float*)d_in[7];
    const float* nb1  = (const float*)d_in[8];
    const float* nw2  = (const float*)d_in[9];
    const float* nb2  = (const float*)d_in[10];
    const float* c0w1 = (const float*)d_in[11];
    const float* c0b1 = (const float*)d_in[12];
    const float* c0w2 = (const float*)d_in[13];
    const float* c0b2 = (const float*)d_in[14];
    const float* c0sw = (const float*)d_in[15];
    const float* c1w1 = (const float*)d_in[16];
    const float* c1b1 = (const float*)d_in[17];
    const float* c1w2 = (const float*)d_in[18];
    const float* c1b2 = (const float*)d_in[19];
    const float* c1sw = (const float*)d_in[20];
    const float* c2w1 = (const float*)d_in[21];
    const float* c2b1 = (const float*)d_in[22];
    const float* c2w2 = (const float*)d_in[23];
    const float* c2b2 = (const float*)d_in[24];
    const float* c2sw = (const float*)d_in[25];

    // ---- workspace carve-up ----
    char* ws = (char*)d_ws;
    unsigned short* wb0 = (unsigned short*)ws;              // 128*256
    unsigned short* wb1 = wb0 + 128 * 256;                  // 128*512
    unsigned short* wb2 = wb1 + 128 * 512;                  // 128*1024
    size_t off = ((size_t)(128 * 256 + 128 * 512 + 128 * 1024) * 2 + 255) & ~(size_t)255;
    float* feat = (float*)(ws + off);                       // [B,128,8400]
    off += (size_t)BATCH * EMB * HWT * sizeof(float);
    float* mx   = (float*)(ws + off); off += (size_t)3 * BATCH * EMB * sizeof(float);
    float* av   = (float*)(ws + off); off += (size_t)3 * BATCH * EMB * sizeof(float);
    float* gate = (float*)(ws + off); off += (size_t)3 * BATCH * EMB * sizeof(float);
    float* smx  = (float*)(ws + off); off += (size_t)BATCH * HWT * sizeof(float);
    float* sav  = (float*)(ws + off); off += (size_t)BATCH * HWT * sizeof(float);
    float* att  = (float*)(ws + off); off += (size_t)BATCH * HWT * sizeof(float);

    // ---- 1) weights f32 -> bf16 ----
    cvt_bf16_kernel<<<(128 * 256 + 255) / 256, 256, 0, stream>>>(nw0, wb0, 128 * 256);
    cvt_bf16_kernel<<<(128 * 512 + 255) / 256, 256, 0, stream>>>(nw1, wb1, 128 * 512);
    cvt_bf16_kernel<<<(128 * 1024 + 255) / 256, 256, 0, stream>>>(nw2, wb2, 128 * 1024);

    // ---- 2) neck GEMMs (WMMA) ----
    neck_wmma_kernel<<<dim3(6400 / NT, BATCH), 256, 0, stream>>>(
        xin0, wb0, nb0, feat, 256, 6400, 0);
    neck_wmma_kernel<<<dim3(1600 / NT, BATCH), 256, 0, stream>>>(
        xin1, wb1, nb1, feat, 512, 1600, 6400);
    neck_wmma_kernel<<<dim3(400 / NT, BATCH), 256, 0, stream>>>(
        xin2, wb2, nb2, feat, 1024, 400, 8000);

    // ---- 3) channel stats ----
    chan_stats_kernel<<<dim3(EMB, BATCH, 3), 256, 0, stream>>>(feat, mx, av);

    // ---- 4) CAM gates ----
    cam_gate_kernel<<<BATCH, EMB, 0, stream>>>(mx, av, c0w1, c0b1, c0w2, c0b2, gate);
    cam_gate_kernel<<<BATCH, EMB, 0, stream>>>(mx + (size_t)BATCH * EMB,
                                               av + (size_t)BATCH * EMB,
                                               c1w1, c1b1, c1w2, c1b2,
                                               gate + (size_t)BATCH * EMB);
    cam_gate_kernel<<<BATCH, EMB, 0, stream>>>(mx + (size_t)2 * BATCH * EMB,
                                               av + (size_t)2 * BATCH * EMB,
                                               c2w1, c2b1, c2w2, c2b2,
                                               gate + (size_t)2 * BATCH * EMB);

    // ---- 5) spatial stats over gated features ----
    spatial_stats_kernel<<<(BATCH * HWT + 255) / 256, 256, 0, stream>>>(feat, gate, smx, sav);

    // ---- 6) SAM 7x7 conv + sigmoid per level ----
    sam_conv_kernel<<<(BATCH * 6400 + 255) / 256, 256, 0, stream>>>(smx, sav, c0sw, att, 80, 80, 0);
    sam_conv_kernel<<<(BATCH * 1600 + 255) / 256, 256, 0, stream>>>(smx, sav, c1sw, att, 40, 40, 6400);
    sam_conv_kernel<<<(BATCH * 400 + 255) / 256, 256, 0, stream>>>(smx, sav, c2sw, att, 20, 20, 8000);

    // ---- 7) outputs: yolo passthrough + gathered embeddings ----
    hipMemcpyAsync(d_out, yolo, (size_t)BATCH * HWT * 6 * sizeof(float),
                   hipMemcpyDeviceToDevice, stream);
    float* emb_out = (float*)d_out + (size_t)BATCH * HWT * 6;
    gather_emb_kernel<<<BATCH * NIDX, EMB, 0, stream>>>(feat, gate, att, ridx, emb_out);
}
